// PackingEnergy_2628519985579
// MI455X (gfx1250) — matrix operand order repaired and checked by reading, not compiled, not run on hardware
//
#include <hip/hip_runtime.h>
#include <hip/hip_fp16.h>

typedef __attribute__((ext_vector_type(16))) _Float16 v16h;
typedef __attribute__((ext_vector_type(8)))  float    v8f;

#define B_    8
#define L_    2048
#define K_    64
#define EMB_  16
#define H1_   128
#define H2_   128
#define LK_   (L_ * K_)          // 131072 pairs per batch
#define NPAIR (B_ * LK_)         // 1,048,576 total pair rows

// padded LDS leading dims (multiples of 8 halfs for 16B alignment; chosen so
// row stride mod 64 banks spreads lanes 0..15 over distinct banks)
#define LDX  72                   // sX rows (64 cols used)
#define LDH  136                  // sH1/sH2 rows (128 cols used)

// ============================================================================
// Kernel 0: zero the output accumulator (harness poisons d_out with 0xAA)
// ============================================================================
__global__ void zero_out_kernel(float* __restrict__ out, int n) {
    int i = blockIdx.x * blockDim.x + threadIdx.x;
    if (i < n) out[i] = 0.0f;
}

// ============================================================================
// Kernel 0b: one-time weight pack. Transposed ([n][k]) f16 images, zero-padded:
//   W1t: 128 x 64   (k 48..63 zeroed)     16 KB
//   W2t: 128 x 128                        32 KB
//   W3t:  16 x 128  (n 7..15 zeroed)       4 KB
// A lane's B-fragment (16 k-values of one column) becomes 32 contiguous bytes.
// ============================================================================
__global__ __launch_bounds__(256) void prep_weights_kernel(
    const float* __restrict__ W1, const float* __restrict__ W2,
    const float* __restrict__ W3,
    _Float16* __restrict__ W1t, _Float16* __restrict__ W2t,
    _Float16* __restrict__ W3t)
{
    const int t = blockIdx.x * blockDim.x + threadIdx.x;
    const int stride = gridDim.x * blockDim.x;
    for (int idx = t; idx < 128 * 64; idx += stride) {
        int n = idx >> 6, k = idx & 63;
        W1t[idx] = (k < 48) ? (_Float16)W1[k * H1_ + n] : (_Float16)0.0f;
    }
    for (int idx = t; idx < 128 * 128; idx += stride) {
        int n = idx >> 7, k = idx & 127;
        W2t[idx] = (_Float16)W2[k * H2_ + n];
    }
    for (int idx = t; idx < 16 * 128; idx += stride) {
        int n = idx >> 7, k = idx & 127;
        W3t[idx] = (n < 7) ? (_Float16)W3[k * 7 + n] : (_Float16)0.0f;
    }
}

// ============================================================================
// Kernel 1: top-K (K=64) nearest non-bonded neighbors per residue.
// One block = 64 residues of one batch. All coords of the batch live in LDS;
// each thread keeps a size-64 max-heap (on squared distance) in LDS,
// node-major so heap slot accesses are bank-conflict free.
// Order of the K results is irrelevant: the consumer sums over K.
// ============================================================================
__global__ __launch_bounds__(64) void topk_kernel(const float* __restrict__ R,
                                                  int*   __restrict__ nbrJ,
                                                  float* __restrict__ nbrR)
{
    __shared__ float sRx[L_], sRy[L_], sRz[L_];          // 24 KB
    __shared__ unsigned long long heap[K_ * 64];          // 32 KB, [node][tid]

    const int tid   = threadIdx.x;
    const int blk   = blockIdx.x;
    const int b     = blk >> 5;          // L_/64 == 32 row-tiles per batch
    const int itile = blk & 31;
    const int i     = itile * 64 + tid;

    const float* Rb = R + (size_t)b * L_ * 3;
    for (int idx = tid; idx < L_; idx += 64) {
        sRx[idx] = Rb[idx * 3 + 0];
        sRy[idx] = Rb[idx * 3 + 1];
        sRz[idx] = Rb[idx * 3 + 2];
    }
#pragma unroll 4
    for (int n = 0; n < K_; ++n) heap[n * 64 + tid] = 0xFFFFFFFFFFFFFFFFull;
    __syncthreads();

    const float xi = sRx[i], yi = sRy[i], zi = sRz[i];

    for (int j = 0; j < L_; ++j) {
        int dij = i - j; dij = dij < 0 ? -dij : dij;
        if (dij <= 3) continue;                           // bonded exclusion (incl. self)
        float dx = xi - sRx[j];
        float dy = yi - sRy[j];
        float dz = zi - sRz[j];
        float d2 = dx * dx + dy * dy + dz * dz;
        // positive-float bits are order-preserving as unsigned; pack j low.
        unsigned long long key =
            ((unsigned long long)__float_as_uint(d2) << 32) | (unsigned)j;
        if (key < heap[0 * 64 + tid]) {                   // below current 64th smallest
            int node = 0;
            for (;;) {                                    // replace-root sift-down
                int l = 2 * node + 1, r = 2 * node + 2;
                int big = node;
                unsigned long long kb = key;
                if (l < K_) { unsigned long long kl = heap[l * 64 + tid]; if (kl > kb) { big = l; kb = kl; } }
                if (r < K_) { unsigned long long kr = heap[r * 64 + tid]; if (kr > kb) { big = r; kb = kr; } }
                if (big == node) { heap[node * 64 + tid] = key; break; }
                heap[node * 64 + tid] = kb;
                node = big;
            }
        }
    }

    const int basep = (b * L_ + i) * K_;
#pragma unroll 4
    for (int k = 0; k < K_; ++k) {
        unsigned long long e = heap[k * 64 + tid];
        int   j  = (int)(e & 0xFFFFFFFFull);
        float d2 = __uint_as_float((unsigned)(e >> 32));
        nbrJ[basep + k] = j;
        nbrR[basep + k] = sqrtf(d2 + 1e-12f);
    }
}

// ============================================================================
// WMMA fragment helpers (CDNA5 wave32 16x16x32 f16; A and B share a contiguous k order)
// A 16x32: lanes 0-15 rows M, lanes 16-31 same rows / upper K-half of group;
//          element h -> k = 16*(lane/16) + h  (one contiguous
//          16-half run -> 2x ds_load_b128)
// B 32x16: lane%16 = column N; lanes 0-15 hold K 0..15, lanes 16-31 K 16..31.
//          Weights are pre-transposed ([n][k]) so the 16 halfs per lane are
//          contiguous -> 2x global_load_b128.
// C/D 16x16 f32: lane%16 = N; VGPR v -> M = v + 8*(lane/16)
// ============================================================================
__device__ inline v16h load_A16(const _Float16* S, int ld, int m0, int k0, int lane) {
    const int half = (lane >> 4) & 1;
    const int m    = m0 + (lane & 15);
    v16h a;
#pragma unroll
    for (int h = 0; h < 16; ++h) {
        int k = (half << 4) + h;
        a[h] = S[m * ld + k0 + k];
    }
    return a;
}

// B fragment from a pre-transposed f16 weight image (global, L2-hot)
__device__ inline v16h load_Bt(const _Float16* __restrict__ W, int ldk, int k0, int n0, int lane) {
    const int kh = ((lane >> 4) & 1) << 4;
    const int n  = n0 + (lane & 15);
    const _Float16* p = W + n * ldk + k0 + kh;      // 16 contiguous halfs (32 B)
    v16h b;
#pragma unroll
    for (int h = 0; h < 16; ++h) b[h] = p[h];
    return b;
}

__device__ inline void store_relu16(_Float16* S, int ld, int m0, int n0, int lane,
                                    v8f c, float bias) {
    const int half = (lane >> 4) & 1;
    const int n    = n0 + (lane & 15);
#pragma unroll
    for (int v = 0; v < 8; ++v) {
        float x = c[v] + bias;
        x = fmaxf(x, 0.0f);
        S[(m0 + v + 8 * half) * ld + n] = (_Float16)x;
    }
}

#define WMMA_F16(a, b, c) \
    __builtin_amdgcn_wmma_f32_16x16x32_f16(false, (a), false, (b), (short)0, (c), false, false)

// ============================================================================
// Kernel 2: fused pair-feature build + 48->128->128->7 MLP + RBF/switch
// epilogue. One block = 64 pair rows, 256 threads = 8 wave32s.
// Wave w owns output column-tile w; loops are k-step-outer with 4 row-tile
// accumulators so every B fragment is fetched exactly once per wave.
// Activations stay in LDS (padded rows -> conflict-free ds_load_b128).
// ============================================================================
__global__ __launch_bounds__(256) void mlp_kernel(
    const int*   __restrict__ seq,     const float* __restrict__ emb,
    const float* __restrict__ b1,      const float* __restrict__ b2,
    const float* __restrict__ b3,
    const float* __restrict__ centers, const float* __restrict__ widths,
    const _Float16* __restrict__ W1t,  const _Float16* __restrict__ W2t,
    const _Float16* __restrict__ W3t,
    const int*   __restrict__ nbrJ,    const float* __restrict__ nbrR,
    float* __restrict__ out)
{
    __shared__ _Float16 sX [64 * LDX];   //  9.0 KB pair features (48 cols, pad 64)
    __shared__ _Float16 sH1[64 * LDH];   // 17.0 KB
    __shared__ _Float16 sH2[64 * LDH];   // 17.0 KB
    __shared__ float    sEmb[20 * EMB_]; //  1.25 KB embedding table
                                         //  total ~44 KB
    const int tid  = threadIdx.x;
    const int lane = tid & 31;
    const int wave = tid >> 5;
    const int base = blockIdx.x * 64;        // first pair row of this block
    const int b    = base / LK_;             // whole block lies in one batch

    // Keep weight images (52 KB f16) hot: 256 B prefetch per thread.
    __builtin_prefetch(&W2t[(tid & 127) * 128], 0, 0);
    __builtin_prefetch(&W1t[(tid & 63) * 128], 0, 0);

    // ---- async-copy the embedding table into LDS (CDNA5 ASYNCcnt path) ----
    if (tid < 80) {                          // 80 x 16 B = 1280 B
        unsigned    l = (unsigned)(size_t)&sEmb[tid * 4];
        const float* g = emb + tid * 4;
        asm volatile("global_load_async_to_lds_b128 %0, %1, off"
                     :: "v"(l), "v"(g) : "memory");
    }
    asm volatile("s_wait_asynccnt 0x0" ::: "memory");

    // ---- build pair features X = [e_i | e_j | e_i*e_j | 0] ----
    int r = tid & 63, g4 = tid >> 6;         // 4 column-groups of 16
    {
        int p   = base + r;
        int rem = p % LK_;
        int i   = rem >> 6;                  // / K_
        int j   = nbrJ[p];
        const int si = seq[b * L_ + i];
        const int sj = seq[b * L_ + j];
        __syncthreads();                     // sEmb ready block-wide
        const float* ei = sEmb + si * EMB_;
        const float* ej = sEmb + sj * EMB_;
        _Float16* xrow = sX + r * LDX + g4 * 16;
#pragma unroll
        for (int c = 0; c < 16; ++c) {
            float v;
            if      (g4 == 0) v = ei[c];
            else if (g4 == 1) v = ej[c];
            else if (g4 == 2) v = ei[c] * ej[c];
            else              v = 0.0f;
            xrow[c] = (_Float16)v;
        }
    }
    __syncthreads();

    const int n0 = wave * 16;

    // ---- layer 1: H1(64x128) = relu(X(64x64) @ W1(64x128) + b1) ----
    {
        const float bv = b1[n0 + (lane & 15)];
        v8f c0 = {}, c1 = {}, c2 = {}, c3 = {};
#pragma unroll
        for (int ks = 0; ks < 2; ++ks) {
            v16h bf = load_Bt(W1t, 64, ks * 32, n0, lane);    // once per k-step
            c0 = WMMA_F16(load_A16(sX, LDX,  0, ks * 32, lane), bf, c0);
            c1 = WMMA_F16(load_A16(sX, LDX, 16, ks * 32, lane), bf, c1);
            c2 = WMMA_F16(load_A16(sX, LDX, 32, ks * 32, lane), bf, c2);
            c3 = WMMA_F16(load_A16(sX, LDX, 48, ks * 32, lane), bf, c3);
        }
        store_relu16(sH1, LDH,  0, n0, lane, c0, bv);
        store_relu16(sH1, LDH, 16, n0, lane, c1, bv);
        store_relu16(sH1, LDH, 32, n0, lane, c2, bv);
        store_relu16(sH1, LDH, 48, n0, lane, c3, bv);
    }
    __syncthreads();

    // ---- layer 2: H2(64x128) = relu(H1 @ W2(128x128) + b2) ----
    {
        const float bv = b2[n0 + (lane & 15)];
        v8f c0 = {}, c1 = {}, c2 = {}, c3 = {};
#pragma unroll
        for (int ks = 0; ks < 4; ++ks) {
            v16h bf = load_Bt(W2t, 128, ks * 32, n0, lane);   // 4 frags total/wave
            c0 = WMMA_F16(load_A16(sH1, LDH,  0, ks * 32, lane), bf, c0);
            c1 = WMMA_F16(load_A16(sH1, LDH, 16, ks * 32, lane), bf, c1);
            c2 = WMMA_F16(load_A16(sH1, LDH, 32, ks * 32, lane), bf, c2);
            c3 = WMMA_F16(load_A16(sH1, LDH, 48, ks * 32, lane), bf, c3);
        }
        store_relu16(sH2, LDH,  0, n0, lane, c0, bv);
        store_relu16(sH2, LDH, 16, n0, lane, c1, bv);
        store_relu16(sH2, LDH, 32, n0, lane, c2, bv);
        store_relu16(sH2, LDH, 48, n0, lane, c3, bv);
    }
    __syncthreads();

    // ---- layer 3 + fused epilogue: softplus, RBF, smooth switch, reduce ----
    float acc = 0.0f;
    if (wave < 4) {                          // 4 row-tiles, one per wave
        const int rt = wave;
        v8f c = {};
#pragma unroll
        for (int ks = 0; ks < 4; ++ks) {
            v16h a  = load_A16(sH2, LDH, rt * 16, ks * 32, lane);
            v16h bf = load_Bt(W3t, 128, ks * 32, 0, lane);
            c = WMMA_F16(a, bf, c);
        }
        const int n    = lane & 15;
        const int half = (lane >> 4) & 1;
        if (n < 7) {
            const float bb3 = b3[n];
            const float cn  = centers[n];
            const float wd  = widths[n];
            const float inv2w2 = 1.0f / (2.0f * wd * wd);
#pragma unroll
            for (int v = 0; v < 8; ++v) {
                int m = rt * 16 + v + 8 * half;          // row within block
                float x = c[v] + bb3;
                float w = (x > 20.0f) ? x : log1pf(expf(x));   // softplus
                float rr = nbrR[base + m];
                float d = rr - cn;
                float phi = expf(-d * d * inv2w2);
                float t = (rr - 10.0f) * 0.5f;           // (r - R_ON)/(R_CUT - R_ON)
                t = fminf(fmaxf(t, 0.0f), 1.0f);
                float sw = 1.0f - t * t * (3.0f - 2.0f * t);
                acc += -w * phi * sw;                    // att * sw contribution
            }
        }
    }
    // wave32 reduction + one atomic per contributing wave
#pragma unroll
    for (int off = 16; off > 0; off >>= 1) acc += __shfl_xor(acc, off, 32);
    if (wave < 4 && lane == 0) unsafeAtomicAdd(&out[b], acc);
}

// ============================================================================
// Host launcher
// ============================================================================
extern "C" void kernel_launch(void* const* d_in, const int* in_sizes, int n_in,
                              void* d_out, int out_size, void* d_ws, size_t ws_size,
                              hipStream_t stream) {
    const float* R   = (const float*)d_in[0];
    const int*   seq = (const int*)  d_in[1];
    const float* emb = (const float*)d_in[2];
    const float* W1  = (const float*)d_in[3];
    const float* b1  = (const float*)d_in[4];
    const float* W2  = (const float*)d_in[5];
    const float* b2  = (const float*)d_in[6];
    const float* W3  = (const float*)d_in[7];
    const float* b3  = (const float*)d_in[8];
    const float* cen = (const float*)d_in[9];
    const float* wid = (const float*)d_in[10];
    float* out = (float*)d_out;

    // workspace layout:
    //   [0, 4MB)        nbrJ   int[NPAIR]
    //   [4MB, 8MB)      nbrR   float[NPAIR]
    //   [8MB, ...)      packed f16 weight images W1t (16KB), W2t (32KB), W3t (4KB)
    char* ws = (char*)d_ws;
    int*      nbrJ = (int*)ws;
    float*    nbrR = (float*)(ws + (size_t)NPAIR * sizeof(int));
    _Float16* W1t  = (_Float16*)(ws + (size_t)NPAIR * 8);
    _Float16* W2t  = W1t + 128 * 64;
    _Float16* W3t  = W2t + 128 * 128;

    zero_out_kernel<<<1, 32, 0, stream>>>(out, out_size);
    prep_weights_kernel<<<dim3(32), dim3(256), 0, stream>>>(W1, W2, W3, W1t, W2t, W3t);
    topk_kernel<<<dim3(B_ * (L_ / 64)), dim3(64), 0, stream>>>(R, nbrJ, nbrR);
    mlp_kernel<<<dim3(NPAIR / 64), dim3(256), 0, stream>>>(
        seq, emb, b1, b2, b3, cen, wid, W1t, W2t, W3t, nbrJ, nbrR, out);
}
